// imageCaptionModel_49143015801228
// MI455X (gfx1250) — matrix operand order repaired and checked by reading, not compiled, not run on hardware
//
#include <hip/hip_runtime.h>
#include <math.h>

// ---------------------------------------------------------------------------
// CDNA5 (gfx1250) wave32 WMMA bf16 GEMM pipeline for the GRU caption model.
//  - v_wmma_f32_16x16x32_bf16 for all matmuls (fp32 accumulate)
//  - A-tiles staged to LDS via TDM tensor_load_to_lds (proj/logits) and
//    global_load_async_to_lds_b128 (GRU step)
//  - B-tiles transposed into LDS with v_perm_b32 pair-packed u32 stores
// ---------------------------------------------------------------------------

typedef __attribute__((ext_vector_type(16))) __bf16 v16bf;
typedef __attribute__((ext_vector_type(8)))  float  v8f;
typedef __attribute__((ext_vector_type(4))) unsigned int u32x4_t;
typedef __attribute__((ext_vector_type(8))) int          i32x8_t;
typedef __attribute__((ext_vector_type(4))) int          i32x4_t;

constexpr int kB = 256;
constexpr int kT = 30;
constexpr int kE = 512;
constexpr int kH = 512;
constexpr int kC = 2048;
constexpr int kV = 10000;
constexpr int kD = kH + kE + kH;   // 1536

#define TB    256     // threads per block (8 waves)
#define BK    32
#define ALP   40      // padded LDS row length for A (bf16): 80B = 20 banks
#define BKP   40      // padded LDS row length for B^T (bf16)
#define BKP32 (BKP / 2)

union FragU { uint4 u[2]; v16bf v; };

__device__ __forceinline__ float sigmoidf_(float x) {
  return 1.0f / (1.0f + __expf(-x));
}

// ---- CDNA5 async global->LDS copy (ASYNCcnt) ------------------------------
__device__ __forceinline__ void async_copy_b128(__bf16* lds_dst,
                                                const __bf16* gsrc) {
  unsigned loff = (unsigned)(uintptr_t)lds_dst;  // flat->LDS: addr[31:0]
  asm volatile("global_load_async_to_lds_b128 %0, %1, off"
               :: "v"(loff), "v"(gsrc) : "memory");
}
__device__ __forceinline__ void wait_async0() {
  asm volatile("s_wait_asynccnt 0x0" ::: "memory");
}

// ---- CDNA5 Tensor Data Mover: DMA a [64 rows x 32 cols] bf16 tile to LDS,
// padding each 64B row with 16B (ALP=40 layout). D# per ISA ch.8.
__device__ __forceinline__ void tdm_load_a_tile(unsigned lds_off,
                                                const __bf16* gsrc, int lda) {
  unsigned long long ga = (unsigned long long)(uintptr_t)gsrc;
  u32x4_t g0;
  g0[0] = 1u;                                    // count=1 (valid descriptor)
  g0[1] = lds_off;                               // lds_addr
  g0[2] = (unsigned)(ga & 0xffffffffu);          // global_addr[31:0]
  g0[3] = (unsigned)((ga >> 32) & 0x01ffffffu)   // global_addr[56:32]
        | (2u << 30);                            // type=2 ("image")
  unsigned td0 = (unsigned)lda;                  // tensor_dim0 (elements)
  i32x8_t g1;
  g1[0] = (int)((1u << 16)        // data_size = 2 bytes
              | (1u << 20)        // pad_enable
              | (3u << 22)        // pad_interval: 16 DWORDs (64B)
              | (3u << 25));      // pad_amount: 4 DWORDs (16B)
  g1[1] = (int)((td0 & 0xffffu) << 16);          // tensor_dim0[15:0]
  g1[2] = (int)(((td0 >> 16) & 0xffffu)          // tensor_dim0[31:16]
              | (64u << 16));                    // tensor_dim1[15:0] = 64
  g1[3] = (int)(32u << 16);                      // tile_dim0 = 32
  g1[4] = 64;                                    // tile_dim1 = 64, tile_dim2=0
  g1[5] = (int)td0;                              // tensor_dim0_stride[31:0]
  g1[6] = 0;                                     // stride hi / dim1_stride lo
  g1[7] = 0;
  i32x4_t z4 = {0, 0, 0, 0};
  i32x8_t z8 = {0, 0, 0, 0, 0, 0, 0, 0};
  __builtin_amdgcn_tensor_load_to_lds(g0, g1, z4, z4, z8, 0);
}
__device__ __forceinline__ void wait_tensor0() {
  __builtin_amdgcn_s_wait_tensorcnt(0);
}

// ---- fragment loads (layouts per CDNA5 ISA 7.12.2) ------------------------
__device__ __forceinline__ v16bf frag_a(const __bf16* sA, int m0, int lane) {
  int row = m0 + (lane & 15);
  int klo = (lane >> 4) * 8;
  const __bf16* p = sA + row * ALP + klo;
  FragU f;
  f.u[0] = *(const uint4*)(p);
  f.u[1] = *(const uint4*)(p + 16);
  return f.v;
}
__device__ __forceinline__ v16bf frag_b(const __bf16* sBt, int n0, int lane) {
  int col = n0 + (lane & 15);
  int kb = (lane >> 4) * 8;
  const __bf16* p = sBt + col * BKP + kb;
  FragU f;
  f.u[0] = *(const uint4*)(p);
  f.u[1] = *(const uint4*)(p + 16);
  return f.v;
}

// ---- tile staging ---------------------------------------------------------
// A: [64][32] bf16, one async b128 per thread (GRU path).
__device__ __forceinline__ void stage_a_async(const __bf16* __restrict__ A,
                                              __bf16* sA, int blockM, int k0,
                                              int lda, int tid) {
  int r  = tid >> 2;          // 0..63
  int cg = (tid & 3) * 8;     // 0,8,16,24
  async_copy_b128(&sA[r * ALP + cg],
                  A + (size_t)(blockM + r) * lda + k0 + cg);
}

// B^T fast path, 32x128 tile: 2 K-rows x 8 cols per thread (2x b128),
// v_perm_b32 pair packing -> 8 u32 LDS stores (conflict-free 20-bank stride).
__device__ __forceinline__ void stage_bt_full128(const __bf16* __restrict__ Bm,
                                                 __bf16* sBt, int blockN,
                                                 int k0, int ldb, int tid) {
  int kp = tid >> 4;            // 0..15 -> K rows 2kp, 2kp+1
  int nc = (tid & 15) * 8;      // 0..120
  const __bf16* p = Bm + (size_t)(k0 + 2 * kp) * ldb + blockN + nc;
  uint4 r0 = *(const uint4*)(p);
  uint4 r1 = *(const uint4*)(p + ldb);
  const unsigned w0[4] = {r0.x, r0.y, r0.z, r0.w};
  const unsigned w1[4] = {r1.x, r1.y, r1.z, r1.w};
  unsigned* dst = (unsigned*)sBt;
#pragma unroll
  for (int i = 0; i < 4; ++i) {
    unsigned lo = __builtin_amdgcn_perm(w1[i], w0[i], 0x05040100u);
    unsigned hi = __builtin_amdgcn_perm(w1[i], w0[i], 0x07060302u);
    dst[(nc + 2 * i)     * BKP32 + kp] = lo;
    dst[(nc + 2 * i + 1) * BKP32 + kp] = hi;
  }
}

// B^T guarded path (logits N-tail only; block-uniform selection).
__device__ __forceinline__ void stage_bt_guard128(const __bf16* __restrict__ Bm,
                                                  __bf16* sBt, int blockN,
                                                  int k0, int ldb, int Nb,
                                                  int tid) {
  int kp = tid >> 4;
  int nc = (tid & 15) * 8;
  const ushort* bs = (const ushort*)Bm;
  unsigned* dst = (unsigned*)sBt;
#pragma unroll
  for (int j = 0; j < 8; ++j) {
    int n = blockN + nc + j;
    unsigned lo = 0, hi = 0;
    if (n < Nb) {
      lo = bs[(size_t)(k0 + 2 * kp) * ldb + n];
      hi = bs[(size_t)(k0 + 2 * kp + 1) * ldb + n];
    }
    dst[(nc + j) * BKP32 + kp] = lo | (hi << 16);
  }
}

// B^T fast path, 32x64 tile (GRU): 2 K-rows x 4 cols (2x b64), 4 perm+stores.
__device__ __forceinline__ void stage_bt_full64(const __bf16* __restrict__ Bm,
                                                __bf16* sBt, int blockN,
                                                int k0, int ldb, int tid) {
  int kp = tid >> 4;            // 0..15
  int nc = (tid & 15) * 4;      // 0..60
  const __bf16* p = Bm + (size_t)(k0 + 2 * kp) * ldb + blockN + nc;
  uint2 r0 = *(const uint2*)(p);
  uint2 r1 = *(const uint2*)(p + ldb);
  const unsigned w0[2] = {r0.x, r0.y};
  const unsigned w1[2] = {r1.x, r1.y};
  unsigned* dst = (unsigned*)sBt;
#pragma unroll
  for (int i = 0; i < 2; ++i) {
    unsigned lo = __builtin_amdgcn_perm(w1[i], w0[i], 0x05040100u);
    unsigned hi = __builtin_amdgcn_perm(w1[i], w0[i], 0x07060302u);
    dst[(nc + 2 * i)     * BKP32 + kp] = lo;
    dst[(nc + 2 * i + 1) * BKP32 + kp] = hi;
  }
}

#define V8F_ZERO {0.f, 0.f, 0.f, 0.f, 0.f, 0.f, 0.f, 0.f}

#define WMMA_BF16(a, b, c) \
  __builtin_amdgcn_wmma_f32_16x16x32_bf16(false, (a), false, (b), (short)0, \
                                          (c), false, false)

// ---- shared 64x128 mainloop (proj & logits); wave tile 32x32 --------------
// A-tile via TDM (wave 0 issues, waits TENSORcnt, barrier publishes).
__device__ __forceinline__ void mainloop_64x128(
    const __bf16* __restrict__ A, const __bf16* __restrict__ Bm, int K,
    int lda, int ldb, int Nb, int blockM, int blockN, int tid, __bf16* sA,
    __bf16* sBt, v8f* acc /*[4]*/) {
  const int lane = tid & 31;
  const int wave = tid >> 5;
  const int m0 = (wave & 1) * 32;
  const int n0 = (wave >> 1) * 32;
  const bool full = (blockN + 128) <= Nb;   // block-uniform
  const unsigned saoff = (unsigned)(uintptr_t)sA;
  const __bf16* Arow = A + (size_t)blockM * lda;
  for (int k0 = 0; k0 < K; k0 += BK) {
    if (wave == 0) tdm_load_a_tile(saoff, Arow + k0, lda);
    if (full) stage_bt_full128(Bm, sBt, blockN, k0, ldb, tid);
    else      stage_bt_guard128(Bm, sBt, blockN, k0, ldb, Nb, tid);
    if (wave == 0) wait_tensor0();
    __syncthreads();
    v16bf a0 = frag_a(sA, m0, lane);
    v16bf a1 = frag_a(sA, m0 + 16, lane);
    v16bf b0 = frag_b(sBt, n0, lane);
    v16bf b1 = frag_b(sBt, n0 + 16, lane);
    acc[0] = WMMA_BF16(a0, b0, acc[0]);
    acc[1] = WMMA_BF16(a0, b1, acc[1]);
    acc[2] = WMMA_BF16(a1, b0, acc[2]);
    acc[3] = WMMA_BF16(a1, b1, acc[3]);
    __syncthreads();
  }
}

// ---------------------------------------------------------------------------
// Kernel 1: baseimgfeat = tanh(cnn @ W_in + b_in)   [256,2048]x[2048,512]
// ---------------------------------------------------------------------------
__global__ __launch_bounds__(TB) void wmma_proj_tanh(
    const __bf16* __restrict__ A, const __bf16* __restrict__ Bm,
    const float* __restrict__ bias, __bf16* __restrict__ out, int K) {
  __shared__ __bf16 sA[64 * ALP];
  __shared__ __bf16 sBt[128 * BKP];
  const int tid = threadIdx.x;
  const int lane = tid & 31;
  const int wave = tid >> 5;
  const int m0 = (wave & 1) * 32;
  const int n0 = (wave >> 1) * 32;
  const int blockM = blockIdx.x * 64;
  const int blockN = blockIdx.y * 128;

  v8f acc[4] = {V8F_ZERO, V8F_ZERO, V8F_ZERO, V8F_ZERO};
  mainloop_64x128(A, Bm, K, K, kH, kH, blockM, blockN, tid, sA, sBt, acc);

  const int nlane = lane & 15;
  const int mhi = (lane >> 4) * 8;
#pragma unroll
  for (int ms = 0; ms < 2; ++ms) {
#pragma unroll
    for (int e = 0; e < 8; ++e) {
      int gm = blockM + m0 + 16 * ms + mhi + e;
#pragma unroll
      for (int h = 0; h < 2; ++h) {
        int gn = blockN + n0 + 16 * h + nlane;
        out[(size_t)gm * kH + gn] =
            (__bf16)tanhf(acc[2 * ms + h][e] + bias[gn]);
      }
    }
  }
}

// ---------------------------------------------------------------------------
// Kernel 2: one fused GRU timestep (cat@W_r, cat@W_c, cat@W_u + gating)
// [256,1536]x[1536,512] x3; block 64x64, wave tile 16x32. Async A staging.
// ---------------------------------------------------------------------------
__global__ __launch_bounds__(TB) void wmma_gru_step(
    const __bf16* __restrict__ catm, const __bf16* __restrict__ Wr,
    const __bf16* __restrict__ Wc, const __bf16* __restrict__ Wu,
    const float* __restrict__ br, const float* __restrict__ bc,
    const float* __restrict__ bu, const float* __restrict__ state_old,
    float* __restrict__ state_new, __bf16* __restrict__ states_all, int t) {
  __shared__ __bf16 sA[64 * ALP];
  __shared__ __bf16 sR[64 * BKP];
  __shared__ __bf16 sC[64 * BKP];
  __shared__ __bf16 sU[64 * BKP];
  const int tid = threadIdx.x;
  const int lane = tid & 31;
  const int wave = tid >> 5;
  const int m0 = (wave & 3) * 16;
  const int n0 = (wave >> 2) * 32;
  const int blockM = blockIdx.x * 64;
  const int blockN = blockIdx.y * 64;

  v8f accR[2] = {V8F_ZERO, V8F_ZERO};
  v8f accC[2] = {V8F_ZERO, V8F_ZERO};
  v8f accU[2] = {V8F_ZERO, V8F_ZERO};

  for (int k0 = 0; k0 < kD; k0 += BK) {
    stage_a_async(catm, sA, blockM, k0, kD, tid);
    stage_bt_full64(Wr, sR, blockN, k0, kH, tid);
    stage_bt_full64(Wc, sC, blockN, k0, kH, tid);
    stage_bt_full64(Wu, sU, blockN, k0, kH, tid);
    wait_async0();
    __syncthreads();
    v16bf a = frag_a(sA, m0, lane);
#pragma unroll
    for (int h = 0; h < 2; ++h) {
      v16bf bR = frag_b(sR, n0 + 16 * h, lane);
      v16bf bC = frag_b(sC, n0 + 16 * h, lane);
      v16bf bU = frag_b(sU, n0 + 16 * h, lane);
      accR[h] = WMMA_BF16(a, bR, accR[h]);
      accC[h] = WMMA_BF16(a, bC, accC[h]);
      accU[h] = WMMA_BF16(a, bU, accU[h]);
    }
    __syncthreads();
  }

  const int nlane = lane & 15;
  const int mhi = (lane >> 4) * 8;
#pragma unroll
  for (int e = 0; e < 8; ++e) {
    int gm = blockM + m0 + mhi + e;
#pragma unroll
    for (int h = 0; h < 2; ++h) {
      int gn = blockN + n0 + 16 * h + nlane;
      float gr = sigmoidf_(accR[h][e] + br[gn]);
      float gu = sigmoidf_(accU[h][e] + bu[gn]);
      float rr = tanhf(gr * accC[h][e] + bc[gn]);
      float so = state_old[(size_t)gm * kH + gn];
      float sn = rr * (1.0f - gu) + gu * so;
      state_new[(size_t)gm * kH + gn] = sn;
      states_all[((size_t)t * kB + gm) * kH + gn] = (__bf16)sn;
    }
  }
}

// ---------------------------------------------------------------------------
// Kernel 3: logits = states @ W_out + b_out.  [7680,512]x[512,10000]
// Rows are (t*B + b); output scattered to [B,T,V] fp32.
// ---------------------------------------------------------------------------
__global__ __launch_bounds__(TB) void wmma_logits(
    const __bf16* __restrict__ states, const __bf16* __restrict__ Wo,
    const float* __restrict__ bo, float* __restrict__ out) {
  __shared__ __bf16 sA[64 * ALP];
  __shared__ __bf16 sBt[128 * BKP];
  const int tid = threadIdx.x;
  const int lane = tid & 31;
  const int wave = tid >> 5;
  const int m0 = (wave & 1) * 32;
  const int n0 = (wave >> 1) * 32;
  const int blockM = blockIdx.x * 64;
  const int blockN = blockIdx.y * 128;

  v8f acc[4] = {V8F_ZERO, V8F_ZERO, V8F_ZERO, V8F_ZERO};
  mainloop_64x128(states, Wo, kH, kH, kV, kV, blockM, blockN, tid, sA, sBt,
                  acc);

  const int nlane = lane & 15;
  const int mhi = (lane >> 4) * 8;
  const bool nfull = (blockN + 128) <= kV;   // block-uniform
#pragma unroll
  for (int ms = 0; ms < 2; ++ms) {
#pragma unroll
    for (int e = 0; e < 8; ++e) {
      int gm = blockM + m0 + 16 * ms + mhi + e;  // = t*kB + b
      int tt = gm >> 8;                          // kB == 256
      int bb = gm & 255;
      float* orow = out + ((size_t)bb * kT + tt) * kV;
      if (nfull) {
#pragma unroll
        for (int h = 0; h < 2; ++h) {
          int gn = blockN + n0 + 16 * h + nlane;
          orow[gn] = acc[2 * ms + h][e] + bo[gn];
        }
      } else {
#pragma unroll
        for (int h = 0; h < 2; ++h) {
          int gn = blockN + n0 + 16 * h + nlane;
          if (gn < kV) orow[gn] = acc[2 * ms + h][e] + bo[gn];
        }
      }
    }
  }
}

// ---------------------------------------------------------------------------
// Small helper kernels
// ---------------------------------------------------------------------------
__global__ void f2bf_kernel(const float* __restrict__ in,
                            __bf16* __restrict__ out, int n4) {
  int i = blockIdx.x * blockDim.x + threadIdx.x;   // n4 = n/4
  if (i < n4) {
    float4 v = ((const float4*)in)[i];
    __bf16 o[4] = {(__bf16)v.x, (__bf16)v.y, (__bf16)v.z, (__bf16)v.w};
    ((uint2*)out)[i] = *(const uint2*)o;
  }
}

__global__ void zero_f32_kernel(float* __restrict__ p, int n) {
  int i = blockIdx.x * blockDim.x + threadIdx.x;
  if (i < n) p[i] = 0.0f;
}

// Build cat = [state (f32->bf16) | emb_table[tok] (f32->bf16) | baseimg (bf16)]
__global__ void prep_cat_kernel(const float* __restrict__ state,
                                const float* __restrict__ emb_table,
                                const __bf16* __restrict__ baseimg,
                                const int* __restrict__ xTokens,
                                __bf16* __restrict__ cat, int t) {
  int idx = blockIdx.x * blockDim.x + threadIdx.x;
  if (idx >= kB * kD) return;
  int b = idx / kD;
  int c = idx % kD;
  __bf16 v;
  if (c < kH) {
    v = (__bf16)state[(size_t)b * kH + c];
  } else if (c < kH + kE) {
    int tok = xTokens[b * kT + t];
    v = (__bf16)emb_table[(size_t)tok * kE + (c - kH)];
  } else {
    v = baseimg[(size_t)b * kH + (c - kH - kE)];
  }
  cat[idx] = v;
}

__global__ void copy_state_kernel(const float* __restrict__ s,
                                  float* __restrict__ out) {
  int i = blockIdx.x * blockDim.x + threadIdx.x;
  if (i < kB * kH) out[i] = s[i];
}

// ---------------------------------------------------------------------------
// Host launch
// ---------------------------------------------------------------------------
extern "C" void kernel_launch(void* const* d_in, const int* in_sizes, int n_in,
                              void* d_out, int out_size, void* d_ws,
                              size_t ws_size, hipStream_t stream) {
  const int*   xTokens   = (const int*)d_in[0];
  const float* cnn       = (const float*)d_in[1];
  const float* emb_table = (const float*)d_in[2];
  const float* W_in      = (const float*)d_in[3];
  const float* b_in      = (const float*)d_in[4];
  const float* W_r       = (const float*)d_in[5];
  const float* b_r       = (const float*)d_in[6];
  const float* W_c       = (const float*)d_in[7];
  const float* b_c       = (const float*)d_in[8];
  const float* W_u       = (const float*)d_in[9];
  const float* b_u       = (const float*)d_in[10];
  const float* W_out     = (const float*)d_in[11];
  const float* b_out     = (const float*)d_in[12];
  float* out = (float*)d_out;

  char* ws = (char*)d_ws;
  size_t off = 0;
  auto alloc = [&](size_t bytes) -> void* {
    off = (off + 255) & ~(size_t)255;
    void* p = ws + off;
    off += bytes;
    return p;
  };

  __bf16* cnn_bf     = (__bf16*)alloc((size_t)kB * kC * 2);
  __bf16* Win_bf     = (__bf16*)alloc((size_t)kC * kH * 2);
  __bf16* Wr_bf      = (__bf16*)alloc((size_t)kD * kH * 2);
  __bf16* Wc_bf      = (__bf16*)alloc((size_t)kD * kH * 2);
  __bf16* Wu_bf      = (__bf16*)alloc((size_t)kD * kH * 2);
  __bf16* Wout_bf    = (__bf16*)alloc((size_t)kH * kV * 2);
  __bf16* baseimg    = (__bf16*)alloc((size_t)kB * kH * 2);
  __bf16* cat_bf     = (__bf16*)alloc((size_t)kB * kD * 2);
  float*  state0     = (float*)alloc((size_t)kB * kH * 4);
  float*  state1     = (float*)alloc((size_t)kB * kH * 4);
  __bf16* states_all = (__bf16*)alloc((size_t)kT * kB * kH * 2);

  auto cvt = [&](const float* src, __bf16* dst, int n) {
    int n4 = n / 4;
    f2bf_kernel<<<(n4 + 255) / 256, 256, 0, stream>>>(src, dst, n4);
  };
  cvt(cnn,   cnn_bf,  kB * kC);
  cvt(W_in,  Win_bf,  kC * kH);
  cvt(W_r,   Wr_bf,   kD * kH);
  cvt(W_c,   Wc_bf,   kD * kH);
  cvt(W_u,   Wu_bf,   kD * kH);
  cvt(W_out, Wout_bf, kH * kV);
  zero_f32_kernel<<<(kB * kH + 255) / 256, 256, 0, stream>>>(state0, kB * kH);

  // baseimgfeat = tanh(cnn @ W_in + b_in)
  wmma_proj_tanh<<<dim3(kB / 64, kH / 128), TB, 0, stream>>>(
      cnn_bf, Win_bf, b_in, baseimg, kC);

  float* sbuf[2] = {state0, state1};
  for (int t = 0; t < kT; ++t) {
    prep_cat_kernel<<<(kB * kD + 255) / 256, 256, 0, stream>>>(
        sbuf[t & 1], emb_table, baseimg, xTokens, cat_bf, t);
    wmma_gru_step<<<dim3(kB / 64, kH / 64), TB, 0, stream>>>(
        cat_bf, Wr_bf, Wc_bf, Wu_bf, b_r, b_c, b_u,
        sbuf[t & 1], sbuf[(t + 1) & 1], states_all, t);
  }

  // logits: [T*B, H] x [H, V] -> out[B,T,V]
  wmma_logits<<<dim3((kT * kB) / 64, (kV + 127) / 128), TB, 0, stream>>>(
      states_all, Wout_bf, b_out, out);

  // current_state (final state is in sbuf[kT & 1] == state0 since kT is even)
  copy_state_kernel<<<(kB * kH + 255) / 256, 256, 0, stream>>>(
      sbuf[kT & 1], out + (size_t)kB * kT * kV);
}
